// CompressedSparseAttention_6373731467587
// MI455X (gfx1250) — matrix operand order repaired and checked by reading, not compiled, not run on hardware
//
#include <hip/hip_runtime.h>

typedef __bf16 bf16;
typedef __attribute__((ext_vector_type(16))) __bf16 v16bf;
typedef __attribute__((ext_vector_type(8)))  float  v8f;

#define S_LEN 4096
#define HID   2048
#define NHEAD 16
#define DHEAD 128
#define CBLK  1024      // S/RATIO
#define NEGV  (-1e30f)

#if defined(__AMDGCN__) && __has_builtin(__builtin_amdgcn_tensor_load_to_lds)
#define USE_TDM 1
#else
#define USE_TDM 0
#endif

static __device__ __forceinline__ v8f wmma_bf16(const v16bf& a, const v16bf& b, const v8f& c) {
  return __builtin_amdgcn_wmma_f32_16x16x32_bf16(false, a, false, b, (short)0, c, false, false);
}

#if USE_TDM
typedef unsigned int v4u __attribute__((ext_vector_type(4)));
typedef int          v8i __attribute__((ext_vector_type(8)));
typedef int          v4i __attribute__((ext_vector_type(4)));

// Issue one TDM 2D tile load: rows x cols (bf16) from global (row stride = stride_elems)
// into contiguous LDS at lds_off. D# packed per CDNA5 ISA 8.3/8.4 (count=1, type=2,
// data_size=1 (2 bytes), tile_dim0=cols, tile_dim1=rows, tensor_dim0_stride=stride).
// 6-arg builtin form (clang-23 toolchain): (g0, g1, g2, g3, extra, cpol).
static __device__ __forceinline__ void tdm_load_tile(unsigned lds_off, const bf16* gptr,
                                                     unsigned rows, unsigned cols,
                                                     unsigned stride_elems) {
  unsigned long long ga = (unsigned long long)(uintptr_t)gptr;
  unsigned td0 = stride_elems;          // generous dim0 extent (no OOB clip inside tile)
  unsigned td1 = rows;
  v4u g0;
  g0[0] = 1u;                                             // count=1, user mode
  g0[1] = lds_off;                                        // lds_addr
  g0[2] = (unsigned)(ga & 0xffffffffu);                   // global_addr[31:0]
  g0[3] = (unsigned)((ga >> 32) & 0x01ffffffu) | (2u << 30); // global_addr[56:32] | type=2
  v8i g1;
  g1[0] = 0x10000;                                        // data_size=1 (2B), mask=0
  g1[1] = (int)((td0 & 0xffffu) << 16);                   // tensor_dim0 lo16
  g1[2] = (int)((td0 >> 16) | ((td1 & 0xffffu) << 16));   // tensor_dim0 hi | tensor_dim1 lo
  g1[3] = (int)((td1 >> 16) | (cols << 16));              // tensor_dim1 hi | tile_dim0
  g1[4] = (int)rows;                                      // tile_dim1 (tile_dim2=0)
  g1[5] = (int)stride_elems;                              // tensor_dim0_stride lo32
  g1[6] = 0;                                              // stride hi16 | dim1_stride lo
  g1[7] = 0;
  v4i z4 = {0, 0, 0, 0};
  v8i z8 = {0, 0, 0, 0, 0, 0, 0, 0};
  __builtin_amdgcn_tensor_load_to_lds(g0, g1, z4, z4, z8, 0);
}
#endif

// ---------------------------------------------------------------- fp32 -> bf16
__global__ void cvt_f32_bf16_kernel(const float* __restrict__ in, bf16* __restrict__ out, int n) {
  int i = blockIdx.x * blockDim.x + threadIdx.x;
  if (i < n) out[i] = (bf16)in[i];
}

// ---------------------------------------------------------------- merge halves
__global__ void merge_kernel(const float* __restrict__ a, const float* __restrict__ b,
                             bf16* __restrict__ out, int n) {
  int i = blockIdx.x * blockDim.x + threadIdx.x;
  if (i < n) out[i] = (bf16)(0.5f * (a[i] + b[i]));
}

// ---------------------------------------------------------------- transpose + convert
// in[R][C] (f32) -> out[C][R] (bf16)
__global__ __launch_bounds__(256)
void transpose_f32_bf16_kernel(const float* __restrict__ in, bf16* __restrict__ out,
                               int R, int Ccols) {
  __shared__ float tile[32][33];
  int bx = blockIdx.x * 32;   // column tile
  int by = blockIdx.y * 32;   // row tile
  int tx = threadIdx.x & 31;
  int ty = threadIdx.x >> 5;  // 0..7
#pragma unroll
  for (int i = 0; i < 32; i += 8)
    tile[ty + i][tx] = in[(size_t)(by + ty + i) * Ccols + bx + tx];
  __syncthreads();
#pragma unroll
  for (int i = 0; i < 32; i += 8)
    out[(size_t)(bx + ty + i) * R + by + tx] = (bf16)tile[tx][ty + i];
}

// ---------------------------------------------------------------- WMMA GEMM
// C[M,N] = A[M,K] @ BT[N,K]^T + bias.  128x128 tile per block, 8 waves, wave = 16x128.
// Both LDS tiles are identical contiguous 2D tiles -> staged via TDM when available.
__global__ __launch_bounds__(256)
void gemm_bf16_kernel(const bf16* __restrict__ A, const bf16* __restrict__ BT,
                      const float* __restrict__ bias, float* __restrict__ C,
                      int M, int N, int K) {
  __shared__ __align__(64) bf16 At[128][32];   // [m][k]
  __shared__ __align__(64) bf16 Bt[128][32];   // [n][k]
  int tid  = threadIdx.x;
  int lane = tid & 31;
  int wave = tid >> 5;
  int hi   = (lane >= 16);
  int l15  = lane & 15;
  int m0 = blockIdx.y * 128;
  int n0 = blockIdx.x * 128;

  v8f acc[8] = {};

  for (int k0 = 0; k0 < K; k0 += 32) {
    __syncthreads();
#if USE_TDM
    if (tid < 32) {   // wave 0 drives the Tensor Data Mover
      tdm_load_tile((unsigned)(uintptr_t)&At[0][0], A  + (size_t)m0 * K + k0, 128, 32, (unsigned)K);
      tdm_load_tile((unsigned)(uintptr_t)&Bt[0][0], BT + (size_t)n0 * K + k0, 128, 32, (unsigned)K);
      __builtin_amdgcn_s_wait_tensorcnt(0);
    }
    __syncthreads();
#else
    {
      int r = tid >> 1;
      int q = (tid & 1) * 2;               // uint4 index 0 or 2 within 64B row
      const uint4* sa = (const uint4*)(A + (size_t)(m0 + r) * K + k0);
      uint4* da = (uint4*)&At[r][0];
      da[q] = sa[q]; da[q + 1] = sa[q + 1];
      const uint4* sb = (const uint4*)(BT + (size_t)(n0 + r) * K + k0);
      uint4* db = (uint4*)&Bt[r][0];
      db[q] = sb[q]; db[q + 1] = sb[q + 1];
    }
    __syncthreads();
#endif
    v16bf a;
    {
      int r  = wave * 16 + l15;
      int kb = hi ? 16 : 0;
#pragma unroll
      for (int i = 0; i < 8; ++i) a[i]     = At[r][kb + i];
#pragma unroll
      for (int i = 0; i < 8; ++i) a[8 + i] = At[r][kb + 8 + i];
    }
    int kb2 = hi ? 16 : 0;
#pragma unroll
    for (int t = 0; t < 8; ++t) {
      v16bf b;
      int n = t * 16 + l15;
#pragma unroll
      for (int i = 0; i < 16; ++i) b[i] = Bt[n][kb2 + i];
      acc[t] = wmma_bf16(a, b, acc[t]);
    }
  }

  int mrow = m0 + wave * 16 + (hi ? 8 : 0);
#pragma unroll
  for (int t = 0; t < 8; ++t) {
    int col = n0 + t * 16 + l15;
    float bv = bias ? bias[col] : 0.0f;
#pragma unroll
    for (int r = 0; r < 8; ++r)
      C[(size_t)(mrow + r) * N + col] = acc[t][r] + bv;
  }
}

// ---------------------------------------------------------------- compression
__global__ __launch_bounds__(256)
void compress_kernel(const float* __restrict__ hs, const float* __restrict__ Wc,
                     const float* __restrict__ bc, bf16* __restrict__ ent_b) {
  __shared__ float red[4][256];
  __shared__ float wsm[4];
  int c = blockIdx.x;
  int t = threadIdx.x;
  float p[4] = {0.f, 0.f, 0.f, 0.f};
  for (int j = t; j < HID; j += 256) {
    float w = Wc[j];
#pragma unroll
    for (int r = 0; r < 4; ++r) p[r] += hs[(size_t)(c * 4 + r) * HID + j] * w;
  }
#pragma unroll
  for (int r = 0; r < 4; ++r) red[r][t] = p[r];
  __syncthreads();
  for (int s = 128; s > 0; s >>= 1) {
    if (t < s) {
#pragma unroll
      for (int r = 0; r < 4; ++r) red[r][t] += red[r][t + s];
    }
    __syncthreads();
  }
  if (t == 0) {
    float l[4], mx = -3e38f;
#pragma unroll
    for (int r = 0; r < 4; ++r) { l[r] = red[r][0] + bc[0]; mx = fmaxf(mx, l[r]); }
    float s = 0.f;
#pragma unroll
    for (int r = 0; r < 4; ++r) { l[r] = __expf(l[r] - mx); s += l[r]; }
#pragma unroll
    for (int r = 0; r < 4; ++r) wsm[r] = l[r] / s;
  }
  __syncthreads();
  for (int j = t; j < HID; j += 256) {
    float e = 0.f;
#pragma unroll
    for (int r = 0; r < 4; ++r) e += wsm[r] * hs[(size_t)(c * 4 + r) * HID + j];
    ent_b[(size_t)c * HID + j] = (bf16)e;
  }
}

// ---------------------------------------------------------------- RMSNorm+RoPE
__global__ __launch_bounds__(128)
void norm_rope_kernel(const float* __restrict__ in, const float* __restrict__ w,
                      bf16* __restrict__ out, int pos_div, int pos_mult, int pos_add) {
  __shared__ float red[128];
  __shared__ float xn[128];
  int row = blockIdx.x;
  int t = threadIdx.x;
  float x = in[(size_t)row * 128 + t];
  red[t] = x * x;
  __syncthreads();
  for (int s = 64; s > 0; s >>= 1) { if (t < s) red[t] += red[t + s]; __syncthreads(); }
  float rms = rsqrtf(red[0] * (1.0f / 128.0f) + 1e-6f);
  float v = x * rms * w[t];
  xn[t] = v;
  __syncthreads();
  int pos = (row / pos_div) * pos_mult + pos_add;
  float o = v;
  if (t < 64) {
    int j = t & 31;
    float inv = __powf(10000.0f, -(float)j * (1.0f / 32.0f));
    float ang = (float)pos * inv;
    float sn, cs;
    __sincosf(ang, &sn, &cs);
    float x1 = xn[j], x2 = xn[j + 32];
    o = (t < 32) ? (x1 * cs - x2 * sn) : (x1 * sn + x2 * cs);
  }
  out[(size_t)row * 128 + t] = (bf16)o;
}

// ---------------------------------------------------------------- sparse (top-k) attention
__global__ __launch_bounds__(32)
void sparse_attn_kernel(const bf16* __restrict__ qb, const bf16* __restrict__ ckb,
                        const float* __restrict__ cvf, float* __restrict__ ctx) {
  __shared__ float sc[16][16];
  int s0   = blockIdx.x * 16;
  int h    = blockIdx.y;
  int lane = threadIdx.x;
  int hi   = (lane >= 16);
  int l15  = lane & 15;
  const float scale = 0.08838834764831845f; // 1/sqrt(128)

  v16bf af[4];
  {
    const bf16* qrow = qb + (size_t)(s0 + l15) * HID + h * 128;
#pragma unroll
    for (int kc = 0; kc < 4; ++kc) {
      int base = kc * 32 + (hi ? 16 : 0);
#pragma unroll
      for (int i = 0; i < 8; ++i) af[kc][i]     = qrow[base + i];
#pragma unroll
      for (int i = 0; i < 8; ++i) af[kc][8 + i] = qrow[base + 8 + i];
    }
  }

  float tv[8]; int ti[8];
#pragma unroll
  for (int i = 0; i < 8; ++i) { tv[i] = -3e38f; ti[i] = 0; }

  for (int cb = 0; cb < 64; ++cb) {
    if (cb < 63)
      __builtin_prefetch(ckb + (size_t)((cb + 1) * 16 + l15) * 128, 0, 1);
    v8f acc = {};
    int kb2 = hi ? 16 : 0;
#pragma unroll
    for (int kc = 0; kc < 4; ++kc) {
      v16bf b;
      const bf16* krow = ckb + (size_t)(cb * 16 + l15) * 128 + kc * 32 + kb2;
#pragma unroll
      for (int i = 0; i < 16; ++i) b[i] = krow[i];
      acc = wmma_bf16(af[kc], b, acc);
    }
    int blk_end = (cb * 16 + l15) * 4 + 3;
#pragma unroll
    for (int m = 0; m < 8; ++m) {
      int q = m + (hi ? 8 : 0);
      float v = acc[m] * scale;
      if (blk_end > s0 + q) v = NEGV;
      sc[q][l15] = v;
    }
    __syncthreads();
    if (lane < 16) {
#pragma unroll
      for (int n = 0; n < 16; ++n) {
        float v = sc[lane][n];
        if (v > tv[7]) {
          tv[7] = v; ti[7] = cb * 16 + n;
#pragma unroll
          for (int k = 7; k > 0; --k) {
            if (tv[k] > tv[k - 1]) {
              float tf = tv[k]; tv[k] = tv[k - 1]; tv[k - 1] = tf;
              int   tt = ti[k]; ti[k] = ti[k - 1]; ti[k - 1] = tt;
            }
          }
        }
      }
    }
    __syncthreads();
  }

  if (lane < 16) {
    float mx = tv[0];
    float e[8], s = 0.f;
#pragma unroll
    for (int k = 0; k < 8; ++k) {
      float sel = (tv[k] > NEGV * 0.5f) ? 1.f : 0.f;
      e[k] = __expf(tv[k] - mx) * sel;
      s += e[k];
    }
    float invs = 1.f / fmaxf(s, 1e-9f);
    float* orow = ctx + (size_t)(s0 + lane) * HID + h * 128;
    for (int d0 = 0; d0 < 128; d0 += 4) {
      float4 a = {0.f, 0.f, 0.f, 0.f};
#pragma unroll
      for (int k = 0; k < 8; ++k) {
        const float4 vv = *(const float4*)(cvf + (size_t)ti[k] * 128 + d0);
        float p = e[k] * invs;
        a.x += p * vv.x; a.y += p * vv.y; a.z += p * vv.z; a.w += p * vv.w;
      }
      *(float4*)(orow + d0) = a;
    }
  }
}

// ---------------------------------------------------------------- local window attention
// lvT layout: [h*128+d][s] so P.V B-fragments are contiguous 32B runs.
__global__ __launch_bounds__(32)
void local_attn_kernel(const bf16* __restrict__ qb, const bf16* __restrict__ lkb,
                       const bf16* __restrict__ lvT, float* __restrict__ ctx) {
  __shared__ float scf[16][256];
  __shared__ bf16  pb[16][256];
  int s0   = blockIdx.x * 16;
  int h    = blockIdx.y;
  int lane = threadIdx.x;
  int hi   = (lane >= 16);
  int l15  = lane & 15;
  int c    = s0 >> 7;
  int kp0  = c * 128 - 128;
  const float scale = 0.08838834764831845f;

  v16bf af[4];
  {
    const bf16* qrow = qb + (size_t)(s0 + l15) * HID + h * 128;
#pragma unroll
    for (int kc = 0; kc < 4; ++kc) {
      int base = kc * 32 + (hi ? 16 : 0);
#pragma unroll
      for (int i = 0; i < 8; ++i) af[kc][i]     = qrow[base + i];
#pragma unroll
      for (int i = 0; i < 8; ++i) af[kc][8 + i] = qrow[base + 8 + i];
    }
  }

  for (int jc = 0; jc < 16; ++jc) {
    int kp  = kp0 + jc * 16 + l15;
    int kpc = kp < 0 ? 0 : kp;
    v8f acc = {};
    int kb2 = hi ? 16 : 0;
#pragma unroll
    for (int kc = 0; kc < 4; ++kc) {
      v16bf b;
      const bf16* krow = lkb + (size_t)kpc * HID + h * 128 + kc * 32 + kb2;
#pragma unroll
      for (int i = 0; i < 16; ++i) b[i] = krow[i];
      acc = wmma_bf16(af[kc], b, acc);
    }
#pragma unroll
    for (int m = 0; m < 8; ++m) {
      int q = m + (hi ? 8 : 0);
      int rel = (s0 + q) - kp;
      float v = acc[m] * scale;
      if (kp < 0 || rel < 0 || rel > 128) v = NEGV;
      scf[q][jc * 16 + l15] = v;
    }
  }
  __syncthreads();

  if (lane < 16) {
    float mx = -3e38f;
    for (int j = 0; j < 256; ++j) mx = fmaxf(mx, scf[lane][j]);
    float s = 0.f;
    for (int j = 0; j < 256; ++j) { float e = __expf(scf[lane][j] - mx); s += e; scf[lane][j] = e; }
    float inv = 1.f / s;
    for (int j = 0; j < 256; ++j) pb[lane][j] = (bf16)(scf[lane][j] * inv);
  }
  __syncthreads();

  v8f acc2[8] = {};
  for (int kc = 0; kc < 8; ++kc) {
    v16bf a;
    int base = kc * 32 + (hi ? 16 : 0);
#pragma unroll
    for (int i = 0; i < 8; ++i) a[i]     = pb[l15][base + i];
#pragma unroll
    for (int i = 0; i < 8; ++i) a[8 + i] = pb[l15][base + 8 + i];
    int kb2 = hi ? 16 : 0;
    int start = kp0 + kc * 32 + kb2;
    if (start < 0) start = 0;          // masked region: probs are 0, data irrelevant
#pragma unroll
    for (int dt = 0; dt < 8; ++dt) {
      v16bf b;
      const bf16* vrow = lvT + (size_t)(h * 128 + dt * 16 + l15) * S_LEN + start;
#pragma unroll
      for (int i = 0; i < 16; ++i) b[i] = vrow[i];
      acc2[dt] = wmma_bf16(a, b, acc2[dt]);
    }
  }

  int mrow = s0 + (hi ? 8 : 0);
#pragma unroll
  for (int dt = 0; dt < 8; ++dt)
#pragma unroll
    for (int m = 0; m < 8; ++m)
      ctx[(size_t)(mrow + m) * HID + h * 128 + dt * 16 + l15] = acc2[dt][m];
}

// ================================================================ host launcher
static inline size_t align256(size_t x) { return (x + 255) & ~(size_t)255; }

extern "C" void kernel_launch(void* const* d_in, const int* in_sizes, int n_in,
                              void* d_out, int out_size, void* d_ws, size_t ws_size,
                              hipStream_t stream) {
  const float* hs   = (const float*)d_in[0];
  const float* Wq   = (const float*)d_in[1];
  const float* bq   = (const float*)d_in[2];
  const float* Wcmp = (const float*)d_in[3];
  const float* bcmp = (const float*)d_in[4];
  const float* Wk   = (const float*)d_in[5];
  const float* bk   = (const float*)d_in[6];
  const float* Wv   = (const float*)d_in[7];
  const float* bv   = (const float*)d_in[8];
  const float* Wlk  = (const float*)d_in[9];
  const float* blk  = (const float*)d_in[10];
  const float* Wlv  = (const float*)d_in[11];
  const float* blv  = (const float*)d_in[12];
  const float* qn_w = (const float*)d_in[13];
  const float* kn_w = (const float*)d_in[14];
  const float* Wo   = (const float*)d_in[15];
  const float* bo   = (const float*)d_in[16];
  float* out = (float*)d_out;

  char* base = (char*)d_ws;
  size_t off = 0;
  auto alloc = [&](size_t bytes) -> void* { void* p = base + off; off = align256(off + bytes); return p; };

  const size_t nSH = (size_t)S_LEN * HID;
  const size_t nCH = (size_t)CBLK * HID;
  const size_t nCD = (size_t)CBLK * DHEAD;

  bf16*  hs_b   = (bf16*)alloc(nSH * 2);
  bf16*  WbigT  = (bf16*)alloc((size_t)HID * HID * 2);   // reused: WqT, WlkT, WlvT, WoT
  bf16*  WsmT   = (bf16*)alloc((size_t)HID * DHEAD * 2); // reused: WkT, WvT
  float* q_f    = (float*)alloc(nSH * 4);                // later reused as ctx_sparse
  float* lk_f   = (float*)alloc(nSH * 4);                // later reused as ctx_local
  float* lv_f   = (float*)alloc(nSH * 4);                // later reused as merged_b
  bf16*  q_b    = (bf16*)alloc(nSH * 2);
  bf16*  lk_b   = (bf16*)alloc(nSH * 2);
  bf16*  lvT_b  = (bf16*)alloc(nSH * 2);                 // lv transposed [h*D+d][s]
  bf16*  ent_b  = (bf16*)alloc(nCH * 2);
  float* ck_f   = (float*)alloc(nCD * 4);
  float* cv_f   = (float*)alloc(nCD * 4);
  bf16*  ck_b   = (bf16*)alloc(nCD * 2);

  float* ctx_sparse = q_f;
  float* ctx_local  = lk_f;
  bf16*  merged_b   = (bf16*)lv_f;

  dim3 blk256(256), blk128(128), blk32(32);
  dim3 gBig(HID / 128, S_LEN / 128);       // 16 x 32
  dim3 gSmall(DHEAD / 128, CBLK / 128);    // 1 x 8
  dim3 gTrW(HID / 32, HID / 32);           // weight transpose 2048x2048
  dim3 gTrWsm(DHEAD / 32, HID / 32);       // 2048x128 -> 128x2048
  dim3 gTrLv(HID / 32, S_LEN / 32);        // 4096x2048 -> 2048x4096

  // 1) activations to bf16
  cvt_f32_bf16_kernel<<<(nSH + 255) / 256, blk256, 0, stream>>>(hs, hs_b, (int)nSH);

  // 2) q / lk / lv projections (transposed weights, shared buffer, stream-serialized)
  transpose_f32_bf16_kernel<<<gTrW, blk256, 0, stream>>>(Wq, WbigT, HID, HID);
  gemm_bf16_kernel<<<gBig, blk256, 0, stream>>>(hs_b, WbigT, bq, q_f, S_LEN, HID, HID);
  transpose_f32_bf16_kernel<<<gTrW, blk256, 0, stream>>>(Wlk, WbigT, HID, HID);
  gemm_bf16_kernel<<<gBig, blk256, 0, stream>>>(hs_b, WbigT, blk, lk_f, S_LEN, HID, HID);
  transpose_f32_bf16_kernel<<<gTrW, blk256, 0, stream>>>(Wlv, WbigT, HID, HID);
  gemm_bf16_kernel<<<gBig, blk256, 0, stream>>>(hs_b, WbigT, blv, lv_f, S_LEN, HID, HID);

  // 3) compression
  compress_kernel<<<CBLK, blk256, 0, stream>>>(hs, Wcmp, bcmp, ent_b);

  // 4) ck / cv projections
  transpose_f32_bf16_kernel<<<gTrWsm, blk256, 0, stream>>>(Wk, WsmT, HID, DHEAD);
  gemm_bf16_kernel<<<gSmall, blk256, 0, stream>>>(ent_b, WsmT, bk, ck_f, CBLK, DHEAD, HID);
  transpose_f32_bf16_kernel<<<gTrWsm, blk256, 0, stream>>>(Wv, WsmT, HID, DHEAD);
  gemm_bf16_kernel<<<gSmall, blk256, 0, stream>>>(ent_b, WsmT, bv, cv_f, CBLK, DHEAD, HID);

  // 5) RMSNorm + RoPE; lv transposed to [h*D+d][s]
  norm_rope_kernel<<<S_LEN * NHEAD, blk128, 0, stream>>>(q_f, qn_w, q_b, NHEAD, 1, 0);
  norm_rope_kernel<<<S_LEN * NHEAD, blk128, 0, stream>>>(lk_f, kn_w, lk_b, NHEAD, 1, 0);
  norm_rope_kernel<<<CBLK, blk128, 0, stream>>>(ck_f, kn_w, ck_b, 1, 4, 3);
  transpose_f32_bf16_kernel<<<gTrLv, blk256, 0, stream>>>(lv_f, lvT_b, S_LEN, HID);

  // 6) sparse top-k attention (writes ctx_sparse, aliases q_f)
  sparse_attn_kernel<<<dim3(S_LEN / 16, NHEAD), blk32, 0, stream>>>(q_b, ck_b, cv_f, ctx_sparse);

  // 7) local sliding-window attention (writes ctx_local, aliases lk_f)
  local_attn_kernel<<<dim3(S_LEN / 16, NHEAD), blk32, 0, stream>>>(q_b, lk_b, lvT_b, ctx_local);

  // 8) merge + output projection
  merge_kernel<<<(nSH + 255) / 256, blk256, 0, stream>>>(ctx_sparse, ctx_local, merged_b, (int)nSH);
  transpose_f32_bf16_kernel<<<gTrW, blk256, 0, stream>>>(Wo, WbigT, HID, HID);
  gemm_bf16_kernel<<<gBig, blk256, 0, stream>>>(merged_b, WbigT, bo, out, S_LEN, HID, HID);

  (void)n_in; (void)in_sizes; (void)out_size; (void)ws_size;
}
